// Dust3RCrossAttention_4973572129471
// MI455X (gfx1250) — matrix operand order repaired and checked by reading, not compiled, not run on hardware
//
#include <hip/hip_runtime.h>

#define DIM   768
#define HEADS 12
#define HDIM  64
#define BATCH 8
#define SEQ   1024
#define BN    (BATCH*SEQ)
#define LDSTR 40   // padded LDS row stride (halves) for 32-wide rows; 16B aligned

typedef _Float16 v8h  __attribute__((ext_vector_type(8)));
typedef _Float16 v16h __attribute__((ext_vector_type(16)));
typedef float    v8f  __attribute__((ext_vector_type(8)));

__device__ __forceinline__ v16h cat8(v8h lo, v8h hi) {
    return __builtin_shufflevector(lo, hi, 0,1,2,3,4,5,6,7,8,9,10,11,12,13,14,15);
}
__device__ __forceinline__ v16h ldfrag(const _Float16* p0, const _Float16* p1) {
    return cat8(*(const v8h*)p0, *(const v8h*)p1);
}
__device__ __forceinline__ v8f wmma_f16(v16h a, v16h b, v8f c) {
    return __builtin_amdgcn_wmma_f32_16x16x32_f16(false, a, false, b, (short)0, c, false, false);
}
// async 16B global->LDS copy (gfx1250, ASYNCcnt-tracked).  LDS byte offset is
// the low 32 bits of the generic pointer.
__device__ __forceinline__ void async_copy16(const void* gptr, void* lptr) {
    asm volatile("global_load_async_to_lds_b128 %0, %1, off"
                 :: "v"((unsigned)(uintptr_t)lptr),
                    "v"((unsigned long long)(uintptr_t)gptr)
                 : "memory");
}

// ---------------------------------------------------------------------------
// Kernel 1: fused Q/K/V projection  out = X @ W^T  (f32 in, f16 out)
// blockIdx.z selects matrix (0=q,1=k,2=v). 128 thr = 4 waves, 64x64 tile.
// q,k stored [b][h][n][d]; v stored transposed [b][h][d][n] for PV matmul.
// ---------------------------------------------------------------------------
__global__ __launch_bounds__(128) void qkv_gemm_kernel(
    const float* __restrict__ q_in, const float* __restrict__ k_in,
    const float* __restrict__ v_in,
    const float* __restrict__ Wq, const float* __restrict__ Wk,
    const float* __restrict__ Wv,
    _Float16* __restrict__ qh, _Float16* __restrict__ kh,
    _Float16* __restrict__ vt)
{
    __shared__ alignas(32) _Float16 lA[64 * LDSTR];
    __shared__ alignas(32) _Float16 lB[64 * LDSTR];

    const int mat = blockIdx.z;
    const float* X = (mat == 0) ? q_in : (mat == 1) ? k_in : v_in;
    const float* W = (mat == 0) ? Wq   : (mat == 1) ? Wk   : Wv;

    const int row0 = blockIdx.y * 64;     // row in [0, BN)
    const int h    = blockIdx.x;          // 64-wide col block == head index
    const int col0 = h * HDIM;

    const int t    = threadIdx.x;
    const int wv   = t >> 5, lane = t & 31;
    const int hg   = lane >> 4, lr = lane & 15;
    const int mrow = wv * 16;

    const int ldr = t >> 1;               // 0..63 (two loaders per row)
    const int ldk = (t & 1) * 16;         // 0 or 16

    v8f acc[4] = {};

    for (int kc = 0; kc < DIM; kc += 32) {
        // stage A (activations) and B (weights, row-major == B^T layout)
        {
            const float4* sa = reinterpret_cast<const float4*>(
                X + (size_t)(row0 + ldr) * DIM + kc + ldk);
            _Float16* da = lA + ldr * LDSTR + ldk;
            const float4* sb = reinterpret_cast<const float4*>(
                W + (size_t)(col0 + ldr) * DIM + kc + ldk);
            _Float16* db = lB + ldr * LDSTR + ldk;
#pragma unroll
            for (int i = 0; i < 4; ++i) {
                float4 fa = sa[i];
                da[4*i+0] = (_Float16)fa.x; da[4*i+1] = (_Float16)fa.y;
                da[4*i+2] = (_Float16)fa.z; da[4*i+3] = (_Float16)fa.w;
                float4 fb = sb[i];
                db[4*i+0] = (_Float16)fb.x; db[4*i+1] = (_Float16)fb.y;
                db[4*i+2] = (_Float16)fb.z; db[4*i+3] = (_Float16)fb.w;
            }
        }
        __syncthreads();

        const _Float16* ap = lA + (mrow + lr) * LDSTR + hg * 16;
        v16h afrag = ldfrag(ap, ap + 8);
#pragma unroll
        for (int ct = 0; ct < 4; ++ct) {
            const _Float16* bp = lB + (ct * 16 + lr) * LDSTR + hg * 16;
            acc[ct] = wmma_f16(afrag, ldfrag(bp, bp + 8), acc[ct]);
        }
        __syncthreads();
    }

    // store: C frag (m = v + 8*hg, n = lr within col tile)
    const int grow    = row0 + mrow + hg * 8;
    const int b       = grow / SEQ;           // 64 | SEQ -> uniform over v
    const int tokb    = grow % SEQ;
    const size_t bhof = (size_t)b * HEADS + h;
#pragma unroll
    for (int ct = 0; ct < 4; ++ct) {
        const int d = ct * 16 + lr;
        if (mat == 2) {
            v8h pk;
#pragma unroll
            for (int v = 0; v < 8; ++v) pk[v] = (_Float16)acc[ct][v];
            *reinterpret_cast<v8h*>(vt + (bhof * HDIM + d) * SEQ + tokb) = pk;
        } else {
            _Float16* dst = ((mat == 0) ? qh : kh)
                          + (bhof * SEQ + tokb) * HDIM + d;
#pragma unroll
            for (int v = 0; v < 8; ++v) dst[(size_t)v * HDIM] = (_Float16)acc[ct][v];
        }
    }
}

// ---------------------------------------------------------------------------
// Kernel 2: RoPE (rotate-half, freq base 100) in-place on q and k (f16)
// one thread per (bh, n, j) pair, j in [0,32)
// ---------------------------------------------------------------------------
__global__ __launch_bounds__(256) void rope_kernel(
    _Float16* __restrict__ qh, _Float16* __restrict__ kh)
{
    const int idx = blockIdx.x * 256 + threadIdx.x;   // BATCH*HEADS*SEQ*32
    const int j   = idx & 31;
    const int n   = (idx >> 5) & (SEQ - 1);
    const int bh  = idx >> 15;                        // / (32*SEQ)

    const float inv = __powf(100.0f, -(float)j * (1.0f / 32.0f));
    float s, c;
    __sincosf((float)n * inv, &s, &c);

    const size_t base = ((size_t)bh * SEQ + n) * HDIM;
    {
        float x1 = (float)qh[base + j], x2 = (float)qh[base + j + 32];
        qh[base + j]      = (_Float16)(x1 * c - x2 * s);
        qh[base + j + 32] = (_Float16)(x2 * c + x1 * s);
    }
    {
        float x1 = (float)kh[base + j], x2 = (float)kh[base + j + 32];
        kh[base + j]      = (_Float16)(x1 * c - x2 * s);
        kh[base + j + 32] = (_Float16)(x2 * c + x1 * s);
    }
}

// ---------------------------------------------------------------------------
// Kernel 3: flash attention. block = 8 waves, one (b,h); each wave owns a
// 16-row Q tile. K/V staged in LDS per 32-key chunk via async-to-LDS copies
// (ASYNCcnt path, shared by all waves); P converted C-layout -> A-layout
// through per-wave LDS scratch.
// ---------------------------------------------------------------------------
__global__ __launch_bounds__(256) void attn_kernel(
    const _Float16* __restrict__ qh, const _Float16* __restrict__ kh,
    const _Float16* __restrict__ vt, _Float16* __restrict__ xo)
{
    __shared__ alignas(32) _Float16 lK[32 * 72];          // 32 keys x 64 d (pad 72)
    __shared__ alignas(32) _Float16 lV[64 * LDSTR];       // 64 d x 32 keys
    __shared__ alignas(32) _Float16 lP[8 * 16 * LDSTR];   // per-wave 16x32 P

    const int t = threadIdx.x, wv = t >> 5, lane = t & 31;
    const int hg = lane >> 4, lr = lane & 15;
    const int bh = blockIdx.y;
    const int t0 = (blockIdx.x * 8 + wv) * 16;            // first query row
    const size_t hbase = (size_t)bh * SEQ * HDIM;

    // Q fragments (A layout) held for the whole pass
    const _Float16* qp = qh + hbase + (size_t)(t0 + lr) * HDIM;
    v16h qa0 = ldfrag(qp + hg * 16,      qp + hg * 16 + 8);
    v16h qa1 = ldfrag(qp + 32 + hg * 16, qp + 32 + hg * 16 + 8);

    v8f o0 = {}, o1 = {}, o2 = {}, o3 = {};
    float M[8], Ls[8];
#pragma unroll
    for (int v = 0; v < 8; ++v) { M[v] = -1e30f; Ls[v] = 0.0f; }

    const _Float16* kbp = kh + hbase;
    const _Float16* vbp = vt + hbase;        // [64][SEQ]
    _Float16* lPw = lP + wv * 16 * LDSTR;
    const float sm = 0.125f;                 // 64^-0.5

    // per-thread staging addresses
    const int kr = t >> 3, ko = (t & 7) * 8;              // K: 32 rows x 64
    const int dr = t >> 2, vo = (t & 3) * 8;              // V: 64 rows x 32

    for (int kt = 0; kt < SEQ; kt += 32) {
        // async staging of K (32x64) and V^T (64x32), 16B per lane
        const _Float16* gk = kbp + (size_t)(kt + kr) * HDIM + ko;
        const _Float16* gv = vbp + (size_t)dr * SEQ + kt + vo;
        async_copy16(gk, lK + kr * 72 + ko);
        async_copy16(gv, lV + dr * LDSTR + vo);
        if (kt + 32 < SEQ) {                 // warm GL2 for next chunk
            __builtin_prefetch(gk + 32 * HDIM, 0, 1);
            __builtin_prefetch(gv + 32, 0, 1);
        }
        asm volatile("s_wait_asynccnt 0x0" ::: "memory");
        __syncthreads();

        // S = Q K^T for two 16-key subtiles (K-dim = 64 -> 2 WMMAs each)
        v8f s0 = {}, s1 = {};
        {
            const _Float16* kp = lK + lr * 72 + hg * 16;
            s0 = wmma_f16(qa0, ldfrag(kp,      kp + 8),  s0);
            s0 = wmma_f16(qa1, ldfrag(kp + 32, kp + 40), s0);
            const _Float16* kq = lK + (16 + lr) * 72 + hg * 16;
            s1 = wmma_f16(qa0, ldfrag(kq,      kq + 8),  s1);
            s1 = wmma_f16(qa1, ldfrag(kq + 32, kq + 40), s1);
        }

        // online softmax (row = v + 8*hg lives across the 16-lane half-group)
        float p0[8], p1[8];
#pragma unroll
        for (int v = 0; v < 8; ++v) {
            float a = s0[v] * sm, b2 = s1[v] * sm;
            float mx = fmaxf(a, b2);
#pragma unroll
            for (int msk = 8; msk >= 1; msk >>= 1)
                mx = fmaxf(mx, __shfl_xor(mx, msk, 32));
            float nM    = fmaxf(M[v], mx);
            float alpha = __expf(M[v] - nM);
            float e0 = __expf(a - nM), e1 = __expf(b2 - nM);
            float rs = e0 + e1;
#pragma unroll
            for (int msk = 8; msk >= 1; msk >>= 1)
                rs += __shfl_xor(rs, msk, 32);
            Ls[v] = Ls[v] * alpha + rs;
            M[v]  = nM;
            p0[v] = e0; p1[v] = e1;
            o0[v] *= alpha; o1[v] *= alpha; o2[v] *= alpha; o3[v] *= alpha;
        }

        // stage P (C layout -> row-major 16x32) in this wave's LDS scratch
#pragma unroll
        for (int v = 0; v < 8; ++v) {
            _Float16* pp = lPw + (v + 8 * hg) * LDSTR;
            pp[lr]      = (_Float16)p0[v];
            pp[16 + lr] = (_Float16)p1[v];
        }
        asm volatile("s_wait_dscnt 0" ::: "memory");   // same-wave LDS RAW

        const _Float16* pa = lPw + lr * LDSTR + hg * 16;
        v16h pf = ldfrag(pa, pa + 8);                  // A frag: 16 q x 32 keys

        const _Float16* vp0 = lV + lr * LDSTR + hg * 16;
        o0 = wmma_f16(pf, ldfrag(vp0,               vp0 + 8),               o0);
        o1 = wmma_f16(pf, ldfrag(vp0 + 16 * LDSTR,  vp0 + 16 * LDSTR + 8),  o1);
        o2 = wmma_f16(pf, ldfrag(vp0 + 32 * LDSTR,  vp0 + 32 * LDSTR + 8),  o2);
        o3 = wmma_f16(pf, ldfrag(vp0 + 48 * LDSTR,  vp0 + 48 * LDSTR + 8),  o3);

        __syncthreads();   // protect lK/lV before next chunk's staging
    }

    // normalize and store to xo: [b][token][h*64 + d] (f16)
    const int b = bh / HEADS, h = bh % HEADS;
#pragma unroll
    for (int v = 0; v < 8; ++v) {
        const int tok = t0 + v + 8 * hg;
        _Float16* dst = xo + ((size_t)b * SEQ + tok) * DIM + h * HDIM;
        const float inv = 1.0f / Ls[v];
        dst[lr]      = (_Float16)(o0[v] * inv);
        dst[16 + lr] = (_Float16)(o1[v] * inv);
        dst[32 + lr] = (_Float16)(o2[v] * inv);
        dst[48 + lr] = (_Float16)(o3[v] * inv);
    }
}

// ---------------------------------------------------------------------------
// Kernel 4: output projection  out = xo @ Wo^T + bo  (f16 x f16 -> f32)
// ---------------------------------------------------------------------------
__global__ __launch_bounds__(128) void out_gemm_kernel(
    const _Float16* __restrict__ xo, const float* __restrict__ Wo,
    const float* __restrict__ bo, float* __restrict__ out)
{
    __shared__ alignas(32) _Float16 lA[64 * LDSTR];
    __shared__ alignas(32) _Float16 lB[64 * LDSTR];

    const int row0 = blockIdx.y * 64, col0 = blockIdx.x * 64;
    const int t = threadIdx.x, wv = t >> 5, lane = t & 31;
    const int hg = lane >> 4, lr = lane & 15;
    const int mrow = wv * 16;
    const int ldr = t >> 1, ldk = (t & 1) * 16;

    v8f acc[4] = {};

    for (int kc = 0; kc < DIM; kc += 32) {
        {
            const _Float16* sa = xo + (size_t)(row0 + ldr) * DIM + kc + ldk;
            *(v8h*)(lA + ldr * LDSTR + ldk)     = *(const v8h*)sa;
            *(v8h*)(lA + ldr * LDSTR + ldk + 8) = *(const v8h*)(sa + 8);
            const float4* sb = reinterpret_cast<const float4*>(
                Wo + (size_t)(col0 + ldr) * DIM + kc + ldk);
            _Float16* db = lB + ldr * LDSTR + ldk;
#pragma unroll
            for (int i = 0; i < 4; ++i) {
                float4 fb = sb[i];
                db[4*i+0] = (_Float16)fb.x; db[4*i+1] = (_Float16)fb.y;
                db[4*i+2] = (_Float16)fb.z; db[4*i+3] = (_Float16)fb.w;
            }
        }
        __syncthreads();

        const _Float16* ap = lA + (mrow + lr) * LDSTR + hg * 16;
        v16h afrag = ldfrag(ap, ap + 8);
#pragma unroll
        for (int ct = 0; ct < 4; ++ct) {
            const _Float16* bp = lB + (ct * 16 + lr) * LDSTR + hg * 16;
            acc[ct] = wmma_f16(afrag, ldfrag(bp, bp + 8), acc[ct]);
        }
        __syncthreads();
    }

#pragma unroll
    for (int ct = 0; ct < 4; ++ct) {
        const int d = col0 + ct * 16 + lr;
        const float bias = bo[d];
#pragma unroll
        for (int v = 0; v < 8; ++v) {
            const int m = row0 + mrow + v + 8 * hg;
            out[(size_t)m * DIM + d] = acc[ct][v] + bias;
        }
    }
}

// ---------------------------------------------------------------------------
extern "C" void kernel_launch(void* const* d_in, const int* in_sizes, int n_in,
                              void* d_out, int out_size, void* d_ws, size_t ws_size,
                              hipStream_t stream)
{
    (void)in_sizes; (void)n_in; (void)out_size; (void)ws_size;
    const float* query = (const float*)d_in[0];
    const float* key_  = (const float*)d_in[1];
    const float* value = (const float*)d_in[2];
    const float* Wq    = (const float*)d_in[3];
    const float* Wk    = (const float*)d_in[4];
    const float* Wv    = (const float*)d_in[5];
    const float* Wo    = (const float*)d_in[6];
    const float* bo    = (const float*)d_in[7];
    float* out = (float*)d_out;

    _Float16* ws = (_Float16*)d_ws;
    const size_t SZ = (size_t)BATCH * HEADS * SEQ * HDIM;   // 6.29M halves
    _Float16* qh = ws;
    _Float16* kh = ws + SZ;
    _Float16* vt = ws + 2 * SZ;
    _Float16* xo = ws + 3 * SZ;   // [B][N][C] f16 attention output

    qkv_gemm_kernel<<<dim3(HEADS, BN / 64, 3), 128, 0, stream>>>(
        query, key_, value, Wq, Wk, Wv, qh, kh, vt);
    rope_kernel<<<dim3((BATCH * HEADS * SEQ * 32) / 256), 256, 0, stream>>>(qh, kh);
    attn_kernel<<<dim3(SEQ / (8 * 16), BATCH * HEADS), 256, 0, stream>>>(
        qh, kh, vt, xo);
    out_gemm_kernel<<<dim3(DIM / 64, BN / 64), 128, 0, stream>>>(xo, Wo, bo, out);
}
